// MCSL_73349451481510
// MI455X (gfx1250) — matrix operand order repaired and checked by reading, not compiled, not run on hardware
//
#include <hip/hip_runtime.h>

#define NN 100000
#define DD 50
#define HH 16
#define KP 64            // K padding for GEMM1 (50 data + 1 bias row + 13 zero)
#define K2 32            // interleaved hi/lo K for GEMM2 (16 pairs)
#define TAUF 0.3f
#define EPSF 1e-20f
#define SLOPE 0.05f

#define A1STEP (HH * KP / 8)   // 128 uint4 per node
#define W2STEP (HH * K2 / 8)   // 64 uint4 per node

typedef __attribute__((ext_vector_type(16))) __bf16 v16bf;
typedef __attribute__((ext_vector_type(8)))  float  v8f;

union FragU {
    v16bf v;
    unsigned short us[16];
    struct { uint4 a, b; } q;
};

struct NodeW {  // one node's pipelined weight fragments
    FragU bh0, bl0, bh1, bl1, w2h, w2l;
};

// Truncation split: f ~= hi + lo, lo compensates hi (pair accuracy ~2^-16).
__device__ __forceinline__ void split2(float f, unsigned short &hi, unsigned short &lo) {
    unsigned u = __float_as_uint(f);
    hi = (unsigned short)(u >> 16);
    float r = f - __uint_as_float(u & 0xFFFF0000u);
    lo = (unsigned short)(__float_as_uint(r) >> 16);
}

__device__ __forceinline__ float leaky(float v) {
    return fmaxf(v, SLOPE * v);   // exact for SLOPE in (0,1)
}

__device__ __forceinline__ v8f wmma_bf16(v16bf a, v16bf b, v8f c) {
    return __builtin_amdgcn_wmma_f32_16x16x32_bf16(
        false, a, false, b, (short)0, c, false, false);
}

// ---------------- Kernel 1: gumbel-sigmoid edge probabilities -> d_out tail
__global__ void mcsl_wprime(const float* __restrict__ w,
                            const float* __restrict__ u1,
                            const float* __restrict__ u2,
                            float* __restrict__ wp) {
    int idx = blockIdx.x * blockDim.x + threadIdx.x;
    if (idx >= DD * DD) return;
    int r = idx / DD, c = idx % DD;
    float g1 = -logf(-logf(u1[idx] + EPSF) + EPSF);
    float g2 = -logf(-logf(u2[idx] + EPSF) + EPSF);
    if (r == c) { g1 = 0.0f; g2 = 0.0f; }
    float logit = (w[idx] + g1 - g2) * (1.0f / TAUF);
    float v = 1.0f / (1.0f + expf(-logit));
    wp[idx] = (r == c) ? 0.0f : v;
}

// ---------------- Kernel 2: bake bf16 weight fragments
__global__ void mcsl_prep(const float* __restrict__ wp,
                          const float* __restrict__ W1,
                          const float* __restrict__ b1,
                          const float* __restrict__ W2,
                          unsigned short* __restrict__ A1hi,
                          unsigned short* __restrict__ A1lo,
                          unsigned short* __restrict__ W2ih,
                          unsigned short* __restrict__ W2il) {
    const int n1 = DD * HH * KP;   // 51200
    const int n2 = DD * HH * K2;   // 25600
    int idx = blockIdx.x * blockDim.x + threadIdx.x;
    if (idx < n1) {
        int i = idx / (HH * KP);
        int rem = idx % (HH * KP);
        int n = rem / KP;
        int k = rem % KP;
        float v = 0.0f;
        if (k < DD)       v = wp[k * DD + i] * W1[(i * DD + k) * HH + n];
        else if (k == DD) v = b1[i * HH + n];   // bias row; x_aug[k==DD]=1
        unsigned short h, l; split2(v, h, l);
        A1hi[idx] = h; A1lo[idx] = l;
    } else if (idx < n1 + n2) {
        int j = idx - n1;
        int i = j / (HH * K2);
        int rem = j % (HH * K2);
        int q = rem / K2;           // k_out (A-operand row)
        int k = rem % K2;           // interleaved K
        int h = k >> 1;
        float v = W2[(i * HH + h) * HH + q];
        unsigned short hh, ll; split2(v, hh, ll);
        W2ih[j] = hh;
        W2il[j] = (k & 1) ? (unsigned short)0 : ll;
    }
}

// ---------------- Kernel 3: main fused MLP, one wave per 16-row tile
__global__ void __launch_bounds__(256) mcsl_main(
    const float* __restrict__ x,
    const unsigned short* __restrict__ A1hi_,
    const unsigned short* __restrict__ A1lo_,
    const unsigned short* __restrict__ W2ih_,
    const unsigned short* __restrict__ W2il_,
    const float* __restrict__ b2_,
    const float* __restrict__ W3_,
    const float* __restrict__ b3,
    float* __restrict__ xhat,
    int numTiles) {
    __shared__ unsigned int h1s[8][16 * 16];   // per-wave packed h1 staging

    const int wave = threadIdx.x >> 5;
    const int lane = threadIdx.x & 31;
    const int tile = blockIdx.x * 8 + wave;
    if (tile >= numTiles) return;

    const int col = lane & 15;
    const int sel = lane >> 4;
    const float* xr = x + (size_t)(tile * 16 + col) * DD;

    // x tile -> A-fragment layout bf16 hi/lo, kept in VGPRs across all 50 nodes.
    v16bf xa_hi[2], xa_lo[2];
#pragma unroll
    for (int kf = 0; kf < 2; ++kf) {
        FragU fh, fl;
#pragma unroll
        for (int half = 0; half < 2; ++half) {
            const int kbase = kf * 32 + sel * 16 + half * 8;
#pragma unroll
            for (int e = 0; e < 8; ++e) {
                const int k = kbase + e;
                float f = (k < DD) ? xr[k] : ((k == DD) ? 1.0f : 0.0f);
                unsigned short h, l; split2(f, h, l);
                fh.us[half * 8 + e] = h;
                fl.us[half * 8 + e] = l;
            }
        }
        xa_hi[kf] = fh.v; xa_lo[kf] = fl.v;
    }

    const uint4*  A1hi4 = (const uint4*)A1hi_;
    const uint4*  A1lo4 = (const uint4*)A1lo_;
    const uint4*  W2ih4 = (const uint4*)W2ih_;
    const uint4*  W2il4 = (const uint4*)W2il_;
    const float4* b2v   = (const float4*)b2_;
    const float4* W3v   = (const float4*)W3_;

    unsigned int* hstore = &h1s[wave][col];
    const unsigned short* hload = (const unsigned short*)h1s[wave] + col * K2 + sel * 8;

    const unsigned a1base = (unsigned)(col * (KP / 8) + sel * 2);
    const unsigned w2base = (unsigned)(col * (K2 / 8));

    auto load_node = [&](NodeW &nw, unsigned a1o, unsigned w2o) {
        nw.bh0.q.a = A1hi4[a1o];      nw.bh0.q.b = A1hi4[a1o + 1];
        nw.bl0.q.a = A1lo4[a1o];      nw.bl0.q.b = A1lo4[a1o + 1];
        nw.bh1.q.a = A1hi4[a1o + 4];  nw.bh1.q.b = A1hi4[a1o + 5];
        nw.bl1.q.a = A1lo4[a1o + 4];  nw.bl1.q.b = A1lo4[a1o + 5];
        nw.w2h.q.a = W2ih4[w2o + sel]; nw.w2h.q.b = W2ih4[w2o + 2 + sel];
        nw.w2l.q.a = W2il4[w2o + sel]; nw.w2l.q.b = W2il4[w2o + 2 + sel];
    };

    auto compute = [&](int i, NodeW &W, unsigned fv) {
        // ---- GEMM1: x_aug @ A1[i] (bias folded into K row 50)
        v8f c = {0.f,0.f,0.f,0.f,0.f,0.f,0.f,0.f};
        c = wmma_bf16(xa_hi[0], W.bh0.v, c);
        c = wmma_bf16(xa_lo[0], W.bh0.v, c);
        c = wmma_bf16(xa_hi[0], W.bl0.v, c);
        c = wmma_bf16(xa_hi[1], W.bh1.v, c);
        c = wmma_bf16(xa_lo[1], W.bh1.v, c);
        c = wmma_bf16(xa_hi[1], W.bl1.v, c);
        // LeakyReLU + pack (hi|lo) -> one b32 LDS store per element
#pragma unroll
        for (int e = 0; e < 8; ++e) {
            float v = leaky(c[e]);
            unsigned vu = __float_as_uint(v);
            float r = v - __uint_as_float(vu & 0xFFFF0000u);
            unsigned word = __builtin_amdgcn_perm(__float_as_uint(r), vu, 0x07060302u);
            hstore[(e + sel * 8) * 16] = word;
        }
        // ---- GEMM2 (transposed, interleaved K=32)
        FragU hB;
        hB.q.a = *(const uint4*)hload;
        hB.q.b = *(const uint4*)(hload + 16);
        const float4 bq0 = b2v[fv];
        const float4 bq1 = b2v[fv + 1];
        v8f c2;
        c2[0] = bq0.x; c2[1] = bq0.y; c2[2] = bq0.z; c2[3] = bq0.w;
        c2[4] = bq1.x; c2[5] = bq1.y; c2[6] = bq1.z; c2[7] = bq1.w;
        c2 = wmma_bf16(W.w2h.v, hB.v, c2);   // Whi*(Hhi+Hlo)
        c2 = wmma_bf16(W.w2l.v, hB.v, c2);   // Wlo*Hhi
        // ---- epilogue: leaky, dot W3, combine k_out halves, store
        const float4 wq0 = W3v[fv];
        const float4 wq1 = W3v[fv + 1];
        float w3v[8] = {wq0.x, wq0.y, wq0.z, wq0.w, wq1.x, wq1.y, wq1.z, wq1.w};
        float acc = 0.0f;
#pragma unroll
        for (int e = 0; e < 8; ++e)
            acc = fmaf(leaky(c2[e]), w3v[e], acc);
        acc += __shfl_xor(acc, 16, 32);
        if (sel == 0) {
            xhat[(size_t)(tile * 16 + col) * DD + i] = acc + b3[i];
        }
    };

    // Software pipeline, two named buffers, rolled 2-node loop (no full unroll).
    NodeW w0, w1;
    unsigned a1 = a1base, w2 = w2base, fv = (unsigned)(sel * 2);
    load_node(w0, a1, w2);
#pragma clang loop unroll(disable)
    for (int i = 0; i < DD - 2; i += 2) {
        load_node(w1, a1 + A1STEP, w2 + W2STEP);
        __builtin_prefetch(&A1hi4[a1 + 2 * A1STEP], 0, 3);   // warm node i+2
        __builtin_prefetch(&A1lo4[a1 + 2 * A1STEP], 0, 3);
        compute(i, w0, fv);
        load_node(w0, a1 + 2 * A1STEP, w2 + 2 * W2STEP);
        compute(i + 1, w1, fv + HH / 4);
        a1 += 2 * A1STEP; w2 += 2 * W2STEP; fv += HH / 2;
    }
    // tail: nodes DD-2, DD-1 (w0 already holds node DD-2)
    load_node(w1, a1 + A1STEP, w2 + W2STEP);
    compute(DD - 2, w0, fv);
    compute(DD - 1, w1, fv + HH / 4);
}

extern "C" void kernel_launch(void* const* d_in, const int* in_sizes, int n_in,
                              void* d_out, int out_size, void* d_ws, size_t ws_size,
                              hipStream_t stream) {
    const float* x  = (const float*)d_in[0];
    const float* w  = (const float*)d_in[1];
    const float* u1 = (const float*)d_in[2];
    const float* u2 = (const float*)d_in[3];
    const float* W1 = (const float*)d_in[4];
    const float* b1 = (const float*)d_in[5];
    const float* W2 = (const float*)d_in[6];
    const float* b2 = (const float*)d_in[7];
    const float* W3 = (const float*)d_in[8];
    const float* b3 = (const float*)d_in[9];

    float* out  = (float*)d_out;
    float* xhat = out;                       // [N, D]
    float* wp   = out + (size_t)NN * DD;     // [D, D] second output

    unsigned short* A1hi = (unsigned short*)d_ws;
    unsigned short* A1lo = A1hi + DD * HH * KP;
    unsigned short* W2h  = A1lo + DD * HH * KP;
    unsigned short* W2l  = W2h + DD * HH * K2;

    mcsl_wprime<<<(DD * DD + 255) / 256, 256, 0, stream>>>(w, u1, u2, wp);

    const int prepTot = DD * HH * KP + DD * HH * K2;
    mcsl_prep<<<(prepTot + 255) / 256, 256, 0, stream>>>(wp, W1, b1, W2,
                                                         A1hi, A1lo, W2h, W2l);

    const int numTiles = NN / 16;  // 6250, exact
    mcsl_main<<<(numTiles + 7) / 8, 256, 0, stream>>>(x, A1hi, A1lo, W2h, W2l,
                                                      b2, W3, b3, xhat, numTiles);
}
